// RelationalNetworkEncoder_27006754357647
// MI455X (gfx1250) — matrix operand order repaired and checked by reading, not compiled, not run on hardware
//
#include <hip/hip_runtime.h>
#include <hip/hip_bf16.h>

typedef __attribute__((ext_vector_type(16))) __bf16 v16bf;
typedef __attribute__((ext_vector_type(8)))  __bf16 v8bf;
typedef __attribute__((ext_vector_type(8)))  float  v8f;

#define CC      256
#define CIN     512
#define HW      196
#define NB      8
#define NPAIRS  (HW * HW)                       // 38416
#define TILE_M  128
#define TILES_PER_B ((NPAIRS + TILE_M - 1) / TILE_M)  // 301
#define LDP     264                             // padded bf16 row stride (bank spread)
#define LDS_TILE (TILE_M * LDP)                 // ushorts per h-tile

__device__ __forceinline__ unsigned short f2bf(float f) {
    unsigned int u = __float_as_uint(f);
    u += 0x7FFFu + ((u >> 16) & 1u);            // round-to-nearest-even
    return (unsigned short)(u >> 16);
}

// A-matrix 16x32 bf16 fragment (ISA 7.12.2): lanes 0-15 hold K = kb+0..7 & kb+16..23,
// lanes 16-31 hold K = kb+8..15 & kb+24..31, for row M = (lane&15) of the subtile.
__device__ __forceinline__ v16bf load_fragA(const unsigned short* base, int row, int stride,
                                            int kb, int lane) {
    int off = row * stride + kb + ((lane >> 4) << 3);
    v8bf lo = *(const v8bf*)(base + off);
    v8bf hi = *(const v8bf*)(base + off + 16);
    return __builtin_shufflevector(lo, hi, 0,1,2,3,4,5,6,7,8,9,10,11,12,13,14,15);
}

// B-matrix 32x16 bf16 fragment: lanes 0-15 hold K = kb+0..7 & kb+16..23, lanes 16-31
// hold K = kb+8..15 & kb+24..31, for column N = (lane&15). Source is w^T stored [N][K].
__device__ __forceinline__ v16bf load_fragB(const unsigned short* base, int row, int stride,
                                            int kb, int lane) {
    int off = row * stride + kb + ((lane >> 4) << 3);
    v8bf lo = *(const v8bf*)(base + off);
    v8bf hi = *(const v8bf*)(base + off + 16);
    return __builtin_shufflevector(lo, hi, 0,1,2,3,4,5,6,7,8,9,10,11,12,13,14,15);
}

__global__ void __launch_bounds__(256) zero_out_kernel(float* out) {
    out[blockIdx.x * CC + threadIdx.x] = 0.0f;
}

// xf[b,p,o] = sum_i im[b,i,p] * proj_w[o,i] + proj_b[o]
__global__ void __launch_bounds__(256) proj_kernel(const float* __restrict__ im,
                                                   const float* __restrict__ pw,
                                                   const float* __restrict__ pb,
                                                   float* __restrict__ xf) {
    __shared__ float row[CIN];
    const int bp = blockIdx.x, b = bp / HW, p = bp - b * HW;
    for (int i = threadIdx.x; i < CIN; i += 256)
        row[i] = im[(b * CIN + i) * HW + p];
    __syncthreads();
    const int o = threadIdx.x;
    float acc = pb[o];
    const float4* w4 = (const float4*)(pw + o * CIN);
    #pragma unroll 4
    for (int i = 0; i < CIN / 4; ++i) {
        float4 w = w4[i];
        acc += w.x * row[4*i] + w.y * row[4*i+1] + w.z * row[4*i+2] + w.w * row[4*i+3];
    }
    xf[bp * CC + o] = acc;
}

// a_l = xf @ w1[:C], a_k = xf @ w1[C:]   (w1 stored [in,out] row-major)
__global__ void __launch_bounds__(256) akl_kernel(const float* __restrict__ xf,
                                                  const float* __restrict__ w1,
                                                  float* __restrict__ al,
                                                  float* __restrict__ ak) {
    __shared__ float row[CC];
    const int bp = blockIdx.x;
    row[threadIdx.x] = xf[bp * CC + threadIdx.x];
    __syncthreads();
    const int o = threadIdx.x;
    float aL = 0.f, aK = 0.f;
    for (int c = 0; c < CC; ++c) {
        float x = row[c];
        aL += x * w1[c * CC + o];
        aK += x * w1[(c + CC) * CC + o];
    }
    al[bp * CC + o] = aL;
    ak[bp * CC + o] = aK;
}

// wT[n*C + k] = bf16(w[k*C + n])
__global__ void __launch_bounds__(256) wtrans_kernel(const float* __restrict__ w,
                                                     unsigned short* __restrict__ wT) {
    const int k = blockIdx.x, n = threadIdx.x;
    wT[n * CC + k] = f2bf(w[k * CC + n]);
}

__global__ void __launch_bounds__(256) rn_pair_kernel(
        const float* __restrict__ a_k, const float* __restrict__ a_l,
        const float* __restrict__ b1,
        const unsigned short* __restrict__ w2T, const float* __restrict__ b2,
        const unsigned short* __restrict__ w3T, const float* __restrict__ b3,
        float* __restrict__ out) {
    extern __shared__ char smem[];
    unsigned short* sh1 = (unsigned short*)smem;
    unsigned short* sh2 = sh1 + LDS_TILE;
    float* colsum = (float*)(smem + (size_t)2 * LDS_TILE * sizeof(unsigned short));

    const int tile = blockIdx.x, b = blockIdx.y;
    const int tid = threadIdx.x, lane = tid & 31, wave = tid >> 5;
    const int rowBase = tile * TILE_M;
    const int validRows = min(TILE_M, NPAIRS - rowBase);

    if (tid < CC) colsum[tid] = 0.0f;

    // ---- Phase 1: h1 = relu(a_k[k] + a_l[l] + b1) -> bf16 LDS tile ----
    for (int idx = tid; idx < TILE_M * (CC / 4); idx += 256) {
        int m = idx >> 6;                 // CC/4 == 64
        int col = (idx & 63) << 2;
        unsigned int lo = 0, hi = 0;
        if (m < validRows) {
            int P = rowBase + m;
            int k = P / HW, l = P - k * HW;
            float4 vk = *(const float4*)(a_k + ((b * HW + k) * CC + col));
            float4 vl = *(const float4*)(a_l + ((b * HW + l) * CC + col));
            float4 vb = *(const float4*)(b1 + col);
            float x0 = fmaxf(vk.x + vl.x + vb.x, 0.f);
            float x1 = fmaxf(vk.y + vl.y + vb.y, 0.f);
            float x2 = fmaxf(vk.z + vl.z + vb.z, 0.f);
            float x3 = fmaxf(vk.w + vl.w + vb.w, 0.f);
            lo = (unsigned int)f2bf(x0) | ((unsigned int)f2bf(x1) << 16);
            hi = (unsigned int)f2bf(x2) | ((unsigned int)f2bf(x3) << 16);
        }
        uint2 st; st.x = lo; st.y = hi;
        *(uint2*)(sh1 + m * LDP + col) = st;
    }
    __syncthreads();

    // Each wave owns 2 N-tiles (32 output columns); full-K B-fragments live in
    // registers (16 x v16bf = 128 VGPRs) and are reused across all 8 M-subtiles.
    const int lr = lane & 15;
    const int mhalf = (lane >> 4) << 3;
    const int n0 = wave * 32 + lr;        // column for nt = 2*wave
    const int n1 = n0 + 16;               // column for nt = 2*wave+1
    v16bf bfr0[8], bfr1[8];

    // ---- Phase 2: h2 = relu(h1 @ w2 + b2) -> bf16 LDS tile ----
    #pragma unroll
    for (int kk = 0; kk < 8; ++kk) {
        bfr0[kk] = load_fragB(w2T, n0, CC, kk * 32, lane);
        bfr1[kk] = load_fragB(w2T, n1, CC, kk * 32, lane);
    }
    const float bias20 = b2[n0], bias21 = b2[n1];

    #pragma unroll 1
    for (int sub = 0; sub < 8; ++sub) {
        v8f acc0 = {}, acc1 = {};
        const int arow = sub * 16 + lr;
        #pragma unroll
        for (int kk = 0; kk < 8; ++kk) {
            v16bf a = load_fragA(sh1, arow, LDP, kk * 32, lane);
            acc0 = __builtin_amdgcn_wmma_f32_16x16x32_bf16(false, a, false, bfr0[kk],
                                                           (short)0, acc0, false, false);
            acc1 = __builtin_amdgcn_wmma_f32_16x16x32_bf16(false, a, false, bfr1[kk],
                                                           (short)0, acc1, false, false);
        }
        const int mrow = sub * 16 + mhalf;
        #pragma unroll
        for (int r = 0; r < 8; ++r) {
            sh2[(mrow + r) * LDP + n0] = f2bf(fmaxf(acc0[r] + bias20, 0.f));
            sh2[(mrow + r) * LDP + n1] = f2bf(fmaxf(acc1[r] + bias21, 0.f));
        }
    }
    __syncthreads();

    // ---- Phase 3: h3 = relu(h2 @ w3 + b3); column-sum valid rows ----
    #pragma unroll
    for (int kk = 0; kk < 8; ++kk) {
        bfr0[kk] = load_fragB(w3T, n0, CC, kk * 32, lane);
        bfr1[kk] = load_fragB(w3T, n1, CC, kk * 32, lane);
    }
    const float bias30 = b3[n0], bias31 = b3[n1];
    float cs0 = 0.f, cs1 = 0.f;

    #pragma unroll 1
    for (int sub = 0; sub < 8; ++sub) {
        v8f acc0 = {}, acc1 = {};
        const int arow = sub * 16 + lr;
        #pragma unroll
        for (int kk = 0; kk < 8; ++kk) {
            v16bf a = load_fragA(sh2, arow, LDP, kk * 32, lane);
            acc0 = __builtin_amdgcn_wmma_f32_16x16x32_bf16(false, a, false, bfr0[kk],
                                                           (short)0, acc0, false, false);
            acc1 = __builtin_amdgcn_wmma_f32_16x16x32_bf16(false, a, false, bfr1[kk],
                                                           (short)0, acc1, false, false);
        }
        const int mbase = sub * 16 + mhalf;
        #pragma unroll
        for (int r = 0; r < 8; ++r) {
            if (mbase + r < validRows) {
                cs0 += fmaxf(acc0[r] + bias30, 0.f);
                cs1 += fmaxf(acc1[r] + bias31, 0.f);
            }
        }
    }
    atomicAdd(&colsum[n0], cs0);
    atomicAdd(&colsum[n1], cs1);
    __syncthreads();

    if (tid < CC) atomicAdd(&out[b * CC + tid], colsum[tid]);
}

extern "C" void kernel_launch(void* const* d_in, const int* in_sizes, int n_in,
                              void* d_out, int out_size, void* d_ws, size_t ws_size,
                              hipStream_t stream) {
    const float* im     = (const float*)d_in[0];
    // d_in[1] boxes, d_in[2] normalized_boxes: unused (deleted by reference)
    const float* proj_w = (const float*)d_in[3];
    const float* proj_b = (const float*)d_in[4];
    const float* w1     = (const float*)d_in[5];
    const float* b1     = (const float*)d_in[6];
    const float* w2     = (const float*)d_in[7];
    const float* b2     = (const float*)d_in[8];
    const float* w3     = (const float*)d_in[9];
    const float* b3     = (const float*)d_in[10];
    float* out = (float*)d_out;

    // workspace carve: xf, a_l, a_k (f32) + w2T, w3T (bf16)
    float* xf = (float*)d_ws;
    float* al = xf + (size_t)NB * HW * CC;
    float* ak = al + (size_t)NB * HW * CC;
    unsigned short* w2T = (unsigned short*)(ak + (size_t)NB * HW * CC);
    unsigned short* w3T = w2T + CC * CC;

    zero_out_kernel<<<dim3(NB), dim3(CC), 0, stream>>>(out);
    proj_kernel<<<dim3(NB * HW), dim3(CC), 0, stream>>>(im, proj_w, proj_b, xf);
    akl_kernel<<<dim3(NB * HW), dim3(CC), 0, stream>>>(xf, w1, al, ak);
    wtrans_kernel<<<dim3(CC), dim3(CC), 0, stream>>>(w2, w2T);
    wtrans_kernel<<<dim3(CC), dim3(CC), 0, stream>>>(w3, w3T);

    size_t ldsBytes = (size_t)2 * LDS_TILE * sizeof(unsigned short) + CC * sizeof(float);
    rn_pair_kernel<<<dim3(TILES_PER_B, NB), dim3(256), ldsBytes, stream>>>(
        ak, al, b1, w2T, b2, w3T, b3, out);
}
